// Heroes_50302656971536
// MI455X (gfx1250) — compile-verified, matrix operand order checked by reading, not run on hardware
//
#include <hip/hip_runtime.h>
#include <hip/hip_bf16.h>

// ---------------------------------------------------------------------------
// Problem constants (from reference)
// ---------------------------------------------------------------------------
#define T_STEPS 200
#define BATCH   2048
#define EDIM    256
#define HDIM    512
#define P0DIM   256
#define P1DIM   128

typedef __attribute__((ext_vector_type(16))) __bf16 v16bf;
typedef __attribute__((ext_vector_type(8)))  __bf16 v8bf;
typedef __attribute__((ext_vector_type(2)))  __bf16 v2bf;
typedef __attribute__((ext_vector_type(8)))  float  v8f;

// fp32 -> bf16 (round to nearest even), raw bits
__device__ __forceinline__ unsigned f32_to_bf16_bits(float f) {
  union { float f; unsigned u; } v; v.f = f;
  unsigned u = v.u;
  return (u + 0x7FFFu + ((u >> 16) & 1u)) >> 16;
}

// pack two fp32 into one dword of two bf16 (lo = a, hi = b)
__device__ __forceinline__ unsigned pack_bf16x2(float a, float b) {
#if __has_builtin(__builtin_amdgcn_cvt_pk_bf16_f32)
  v2bf p = __builtin_amdgcn_cvt_pk_bf16_f32(a, b);  // v_cvt_pk_bf16_f32
  unsigned u;
  __builtin_memcpy(&u, &p, 4);
  return u;
#else
  return f32_to_bf16_bits(a) | (f32_to_bf16_bits(b) << 16);
#endif
}

// Load one WMMA 16-bit fragment (16 halves) from LDS:
//  vgprs 0-3 = halves [p, p+8), vgprs 4-7 = halves [p+16, p+24)
__device__ __forceinline__ v16bf load_frag(const unsigned short* p) {
  v8bf lo = *(const v8bf*)p;        // ds_load_b128
  v8bf hi = *(const v8bf*)(p + 16); // ds_load_b128
  return __builtin_shufflevector(lo, hi, 0,1,2,3,4,5,6,7,8,9,10,11,12,13,14,15);
}

__device__ __forceinline__ float sigmoidf_(float x) {
  return 1.0f / (1.0f + __expf(-x));
}

// ---------------------------------------------------------------------------
// Generic bf16 WMMA GEMM:
//   C[M,N] = act( [A0 | A1] @ Bt^T + bias )
// A0: [M, K0] fp32 row-major (converted to bf16 while staging to LDS)
// A1: [M, K1] fp32 row-major (optional, K1 == 0 to disable)
// Bt: [N, K0+K1] bf16 row-major (pre-transposed / pre-packed)
// RSM0/RSM1 (compile-time): per-row scaling of A segment: 0=none, 1=*g, 2=*(1-g)
// MODE (compile-time): 0=linear, 1=gates(sigmoid if col<3H else tanh),
//                      2=tanh, 3=leaky(0.3)
// HAS_BIAS (compile-time): add bias[col] before activation
// Requirements: M %128==0, N %128==0, K0 %64==0, K1 %64==0.
// ---------------------------------------------------------------------------
template<int MODE, int RSM0, int RSM1, bool HAS_BIAS>
__global__ __launch_bounds__(256)
void wmma_gemm_bf16(const float* __restrict__ A0, int lda0, int K0,
                    const float* __restrict__ A1, int lda1, int K1,
                    const unsigned short* __restrict__ Bt,
                    const float* __restrict__ bias,
                    float* __restrict__ C, int N,
                    const float* __restrict__ gvec)
{
  __shared__ unsigned short sA[128 * 64];
  __shared__ unsigned short sB[128 * 64];

  const int tid   = threadIdx.x;
  const int w     = tid >> 5;
  const int lane  = tid & 31;
  const int wm    = (w & 3) * 32;   // wave M origin within 128 tile
  const int wn    = (w >> 2) * 64;  // wave N origin within 128 tile
  const int tileN = blockIdx.x * 128;
  const int tileM = blockIdx.y * 128;
  const int Ktot  = K0 + K1;

  v8f acc[2][4] = {};

  const int r  = tid >> 1;        // 0..127 : tile row staged by this thread
  const int cb = (tid & 1) * 32;  // 0 or 32: k-chunk within 64-wide stage

  // ---- loop-invariant staging state (hoisted out of the K loop) ----
  const int rowg = tileM + r;
  const float* A0row = A0 + (long)rowg * lda0;
  const float* A1row = A1 + (long)rowg * lda1;
  float sc0 = 1.0f, sc1 = 1.0f;
  if (RSM0 == 1 || RSM0 == 2 || RSM1 == 1 || RSM1 == 2) {
    const float g = gvec[rowg];
    if (RSM0 == 1) sc0 = g; else if (RSM0 == 2) sc0 = 1.0f - g;
    if (RSM1 == 1) sc1 = g; else if (RSM1 == 2) sc1 = 1.0f - g;
  }
  unsigned* dstA = (unsigned*)&sA[r * 64 + cb];
  const unsigned short* Brow = Bt + (long)(tileN + r) * Ktot + cb;

  for (int kk = 0; kk < Ktot; kk += 64) {
    // ---- stage A tile (vectorized fp32 -> packed bf16x2) ----
    {
      const int kg = kk + cb;
      const float* Ap;
      float sc;
      if (kg < K0) { Ap = A0row + kg;        sc = sc0; }
      else         { Ap = A1row + (kg - K0); sc = sc1; }
      #pragma unroll
      for (int j = 0; j < 8; ++j) {
        float4 f = ((const float4*)Ap)[j];
        dstA[2 * j]     = pack_bf16x2(f.x * sc, f.y * sc);
        dstA[2 * j + 1] = pack_bf16x2(f.z * sc, f.w * sc);
      }
    }
    // ---- stage B tile (pre-packed bf16 [N][Ktot]) ----
    {
      const uint4* Bp = (const uint4*)(Brow + kk);
      uint4* d = (uint4*)&sB[r * 64 + cb];
      d[0] = Bp[0]; d[1] = Bp[1]; d[2] = Bp[2]; d[3] = Bp[3];
    }
    __syncthreads();

    #pragma unroll
    for (int ks = 0; ks < 64; ks += 32) {
      const int koff = (lane >> 4) * 8;
      v16bf afr[2], bfr[4];
      #pragma unroll
      for (int m = 0; m < 2; ++m)
        afr[m] = load_frag(&sA[(wm + m * 16 + (lane & 15)) * 64 + ks + koff]);
      #pragma unroll
      for (int n = 0; n < 4; ++n)
        bfr[n] = load_frag(&sB[(wn + n * 16 + (lane & 15)) * 64 + ks + koff]);
      #pragma unroll
      for (int m = 0; m < 2; ++m)
        #pragma unroll
        for (int n = 0; n < 4; ++n)
          acc[m][n] = __builtin_amdgcn_wmma_f32_16x16x32_bf16(
              false, afr[m], false, bfr[n], (short)0, acc[m][n], false, false);
    }
    __syncthreads();
  }

  // ---- epilogue: bias + activation, ISA C-layout scatter ----
  const int cl    = lane & 15;
  const int rhalf = (lane >> 4) * 8;
  #pragma unroll
  for (int n = 0; n < 4; ++n) {
    const int col = tileN + wn + n * 16 + cl;
    float bv = 0.0f;
    if constexpr (HAS_BIAS) bv = bias[col];
    #pragma unroll
    for (int m = 0; m < 2; ++m) {
      const int rowb = tileM + wm + m * 16 + rhalf;
      #pragma unroll
      for (int v = 0; v < 8; ++v) {
        float x = acc[m][n][v] + bv;
        if constexpr (MODE == 1) x = (col < 3 * HDIM) ? sigmoidf_(x) : tanhf(x);
        else if constexpr (MODE == 2) x = tanhf(x);
        else if constexpr (MODE == 3) x = (x > 0.0f) ? x : 0.3f * x;
        C[(long)(rowb + v) * N + col] = x;
      }
    }
  }
}

// ---------------------------------------------------------------------------
// Final dense layer to scalar + sigmoid (+ optional per-row multiplier):
//   p[row] = sigmoid( dot(X[row,:], w) + bias ) * (mult ? mult[row] : 1)
// One wave per row (wave32 reduction).
// ---------------------------------------------------------------------------
__global__ __launch_bounds__(256)
void dot_sigmoid(const float* __restrict__ X, int K,
                 const float* __restrict__ w, const float* __restrict__ bias,
                 const float* __restrict__ mult,
                 float* __restrict__ gout, float* __restrict__ predout)
{
  const int row  = blockIdx.x * 8 + (threadIdx.x >> 5);
  const int lane = threadIdx.x & 31;
  const float* xr = X + (long)row * K;
  float s = 0.0f;
  for (int k = lane; k < K; k += 32) s += xr[k] * w[k];
  #pragma unroll
  for (int off = 16; off > 0; off >>= 1) s += __shfl_xor(s, off, 32);
  if (lane == 0) {
    float p = sigmoidf_(s + bias[0]);
    if (mult) p *= mult[row];
    if (gout) gout[row] = p;
    predout[row] = p;
  }
}

// ---------------------------------------------------------------------------
// State updates (click / conversion streams)
// gates: [B,4H] already activated (f,i,o sigmoid; g tanh)
// ---------------------------------------------------------------------------
__global__ void update_c_kernel(const float* __restrict__ gates,
                                const float* __restrict__ shat,
                                const float* __restrict__ g,
                                float* __restrict__ s_c, float* __restrict__ H_c)
{
  const long i = (long)blockIdx.x * 256 + threadIdx.x;
  if (i >= (long)BATCH * HDIM) return;
  const int b = (int)(i / HDIM), h = (int)(i % HDIM);
  const float* gr = gates + (long)b * 4 * HDIM;
  const float f = gr[h], ii = gr[HDIM + h], o = gr[2 * HDIM + h], gc = gr[3 * HDIM + h];
  const float gb = g[b];
  const float s = shat[i] + ii * gc + (1.0f - gb) * (f * s_c[i]);
  s_c[i] = s;
  H_c[i] = o * tanhf(s);
}

__global__ void update_v_kernel(const float* __restrict__ gates,
                                const float* __restrict__ shat,
                                const float* __restrict__ g,
                                float* __restrict__ s_v, float* __restrict__ H_v)
{
  const long i = (long)blockIdx.x * 256 + threadIdx.x;
  if (i >= (long)BATCH * HDIM) return;
  const int b = (int)(i / HDIM), h = (int)(i % HDIM);
  const float* gr = gates + (long)b * 4 * HDIM;
  const float f = gr[h], ii = gr[HDIM + h], o = gr[2 * HDIM + h], gv = gr[3 * HDIM + h];
  const float gb = g[b];
  const float s_old = s_v[i];
  const float s = shat[i] + (1.0f - gb) * s_old + gb * (f * s_old + ii * gv);
  s_v[i] = s;
  H_v[i] = (1.0f - gb) * H_v[i] + gb * (o * tanhf(s));
}

__global__ void zero_f32(float* __restrict__ p, long n) {
  const long i = (long)blockIdx.x * 256 + threadIdx.x;
  if (i < n) p[i] = 0.0f;
}

// ---------------------------------------------------------------------------
// Weight packing: fp32 -> bf16, transposed to [N][K] for the WMMA B operand
// ---------------------------------------------------------------------------
// Wx [4][E][H], Wh [4][H][H]  ->  Bt [4H][E+H]; n = gate*H + h
__global__ void pack_xh(const float* __restrict__ Wx, const float* __restrict__ Wh,
                        unsigned short* __restrict__ Bt)
{
  const long i = (long)blockIdx.x * 256 + threadIdx.x;
  const int Kt = EDIM + HDIM;
  const long total = (long)4 * HDIM * Kt;
  if (i >= total) return;
  const int n = (int)(i / Kt), k = (int)(i % Kt);
  const int gate = n / HDIM, h = n % HDIM;
  float v;
  if (k < EDIM) v = Wx[((long)gate * EDIM + k) * HDIM + h];
  else          v = Wh[((long)gate * HDIM + (k - EDIM)) * HDIM + h];
  Bt[i] = (unsigned short)f32_to_bf16_bits(v);
}

// Wa [H][H], Wb [H][H] -> Bt [H][2H]
__global__ void pack_pair(const float* __restrict__ Wa, const float* __restrict__ Wb,
                          unsigned short* __restrict__ Bt)
{
  const long i = (long)blockIdx.x * 256 + threadIdx.x;
  const long total = (long)HDIM * 2 * HDIM;
  if (i >= total) return;
  const int n = (int)(i / (2 * HDIM)), k = (int)(i % (2 * HDIM));
  float v = (k < HDIM) ? Wa[(long)k * HDIM + n] : Wb[(long)(k - HDIM) * HDIM + n];
  Bt[i] = (unsigned short)f32_to_bf16_bits(v);
}

// W [Kin][Kout] -> Bt [Kout][Kin]
__global__ void pack_t(const float* __restrict__ W, int Kin, int Kout,
                       unsigned short* __restrict__ Bt)
{
  const long i = (long)blockIdx.x * 256 + threadIdx.x;
  const long total = (long)Kin * Kout;
  if (i >= total) return;
  const int n = (int)(i / Kin), k = (int)(i % Kin);
  Bt[i] = (unsigned short)f32_to_bf16_bits(W[(long)k * Kout + n]);
}

// ---------------------------------------------------------------------------
// Host driver
// ---------------------------------------------------------------------------
extern "C" void kernel_launch(void* const* d_in, const int* in_sizes, int n_in,
                              void* d_out, int out_size, void* d_ws, size_t ws_size,
                              hipStream_t stream)
{
  (void)in_sizes; (void)n_in; (void)out_size; (void)ws_size;

  const float* X    = (const float*)d_in[0];
  const float* Wx_c = (const float*)d_in[1];
  const float* bx_c = (const float*)d_in[2];   // [4,H] flat == bias[n]
  const float* Wh_c = (const float*)d_in[3];
  const float* Wx_v = (const float*)d_in[4];
  const float* bx_v = (const float*)d_in[5];
  const float* Wh_v = (const float*)d_in[6];
  const float* Ws   = (const float*)d_in[7];   // [4,H,H]
  const float* Wpc0 = (const float*)d_in[8];
  const float* bpc0 = (const float*)d_in[9];
  const float* Wpc1 = (const float*)d_in[10];
  const float* bpc1 = (const float*)d_in[11];
  const float* Wfcc = (const float*)d_in[12];  // [P1,1] -> vector of P1
  const float* bfcc = (const float*)d_in[13];
  const float* Wpv0 = (const float*)d_in[14];
  const float* bpv0 = (const float*)d_in[15];
  const float* Wpv1 = (const float*)d_in[16];
  const float* bpv1 = (const float*)d_in[17];
  const float* Wfcv = (const float*)d_in[18];
  const float* bfcv = (const float*)d_in[19];

  float* out    = (float*)d_out;
  float* pred_c = out;
  float* pred_v = out + (long)T_STEPS * BATCH;

  // ---- workspace layout ----
  char* wsb = (char*)d_ws;
  size_t off = 0;
  auto walloc = [&](size_t bytes) -> void* {
    void* p = wsb + off;
    off = (off + bytes + 255) & ~(size_t)255;
    return p;
  };
  unsigned short* BtXHc  = (unsigned short*)walloc((size_t)4 * HDIM * (EDIM + HDIM) * 2);
  unsigned short* BtXHv  = (unsigned short*)walloc((size_t)4 * HDIM * (EDIM + HDIM) * 2);
  unsigned short* BtWs01 = (unsigned short*)walloc((size_t)HDIM * 2 * HDIM * 2);
  unsigned short* BtWs23 = (unsigned short*)walloc((size_t)HDIM * 2 * HDIM * 2);
  unsigned short* BtPc0  = (unsigned short*)walloc((size_t)P0DIM * HDIM * 2);
  unsigned short* BtPc1  = (unsigned short*)walloc((size_t)P1DIM * P0DIM * 2);
  unsigned short* BtPv0  = (unsigned short*)walloc((size_t)P0DIM * HDIM * 2);
  unsigned short* BtPv1  = (unsigned short*)walloc((size_t)P1DIM * P0DIM * 2);
  float* states = (float*)walloc((size_t)4 * BATCH * HDIM * 4);
  float* Hc  = states;
  float* Hv  = states + (long)BATCH * HDIM;
  float* scs = states + (long)2 * BATCH * HDIM;
  float* svs = states + (long)3 * BATCH * HDIM;
  float* gates = (float*)walloc((size_t)BATCH * 4 * HDIM * 4);
  float* shat  = (float*)walloc((size_t)BATCH * HDIM * 4);
  float* p0buf = (float*)walloc((size_t)BATCH * P0DIM * 4);
  float* p1buf = (float*)walloc((size_t)BATCH * P1DIM * 4);
  float* gvec  = (float*)walloc((size_t)BATCH * 4);
  float* dummy = (float*)walloc((size_t)BATCH * 4);

  // ---- pack weights to bf16 [N][K] ----
  {
    long t = (long)4 * HDIM * (EDIM + HDIM);
    pack_xh<<<(unsigned)((t + 255) / 256), 256, 0, stream>>>(Wx_c, Wh_c, BtXHc);
    pack_xh<<<(unsigned)((t + 255) / 256), 256, 0, stream>>>(Wx_v, Wh_v, BtXHv);
  }
  {
    long t = (long)HDIM * 2 * HDIM;
    const long HH = (long)HDIM * HDIM;
    pack_pair<<<(unsigned)((t + 255) / 256), 256, 0, stream>>>(Ws,          Ws + HH,     BtWs01);
    pack_pair<<<(unsigned)((t + 255) / 256), 256, 0, stream>>>(Ws + 2 * HH, Ws + 3 * HH, BtWs23);
  }
  pack_t<<<(HDIM * P0DIM + 255) / 256, 256, 0, stream>>>(Wpc0, HDIM, P0DIM, BtPc0);
  pack_t<<<(P0DIM * P1DIM + 255) / 256, 256, 0, stream>>>(Wpc1, P0DIM, P1DIM, BtPc1);
  pack_t<<<(HDIM * P0DIM + 255) / 256, 256, 0, stream>>>(Wpv0, HDIM, P0DIM, BtPv0);
  pack_t<<<(P0DIM * P1DIM + 255) / 256, 256, 0, stream>>>(Wpv1, P0DIM, P1DIM, BtPv1);

  // ---- zero recurrent state ----
  {
    long t = (long)4 * BATCH * HDIM;
    zero_f32<<<(unsigned)((t + 255) / 256), 256, 0, stream>>>(states, t);
  }

  const dim3 blk(256);
  const dim3 gGates(4 * HDIM / 128, BATCH / 128);  // (16,16)
  const dim3 gShat (HDIM / 128,     BATCH / 128);  // (4,16)
  const dim3 gP0   (P0DIM / 128,    BATCH / 128);  // (2,16)
  const dim3 gP1   (P1DIM / 128,    BATCH / 128);  // (1,16)
  const unsigned gUpd = (unsigned)(((long)BATCH * HDIM + 255) / 256);

  auto predict = [&](const float* Hin,
                     const unsigned short* B0, const float* b0,
                     const unsigned short* B1, const float* b1,
                     const float* wf, const float* bf,
                     const float* mult, float* gout, float* pout) {
    wmma_gemm_bf16<3, 0, 0, true><<<gP0, blk, 0, stream>>>(
        Hin, HDIM, HDIM, nullptr, 0, 0, B0, b0, p0buf, P0DIM, nullptr);
    wmma_gemm_bf16<3, 0, 0, true><<<gP1, blk, 0, stream>>>(
        p0buf, P0DIM, P0DIM, nullptr, 0, 0, B1, b1, p1buf, P1DIM, nullptr);
    dot_sigmoid<<<BATCH / 8, blk, 0, stream>>>(p1buf, P1DIM, wf, bf, mult, gout, pout);
  };

  // g0 = predict_c(H_c = 0)
  predict(Hc, BtPc0, bpc0, BtPc1, bpc1, Wfcc, bfcc, nullptr, gvec, dummy);

  for (int t = 0; t < T_STEPS; ++t) {
    const float* Xt = X + (long)t * BATCH * EDIM;

    // click gates: sigmoid/tanh( [X_t | H_c] @ [Wx_c;Wh_c] + bx_c )
    wmma_gemm_bf16<1, 0, 0, true><<<gGates, blk, 0, stream>>>(
        Xt, EDIM, EDIM, Hc, HDIM, HDIM, BtXHc, bx_c, gates, 4 * HDIM, nullptr);
    // s_c_hat = tanh( [(1-g)H_c | gH_v] @ [Ws0;Ws1] )
    wmma_gemm_bf16<2, 2, 1, false><<<gShat, blk, 0, stream>>>(
        Hc, HDIM, HDIM, Hv, HDIM, HDIM, BtWs01, nullptr, shat, HDIM, gvec);
    update_c_kernel<<<gUpd, blk, 0, stream>>>(gates, shat, gvec, scs, Hc);

    // H_c_p -> g (new), pred_c[t]
    predict(Hc, BtPc0, bpc0, BtPc1, bpc1, Wfcc, bfcc,
            nullptr, gvec, pred_c + (long)t * BATCH);

    // conversion gates (driven by updated H_c)
    wmma_gemm_bf16<1, 0, 0, true><<<gGates, blk, 0, stream>>>(
        Xt, EDIM, EDIM, Hc, HDIM, HDIM, BtXHv, bx_v, gates, 4 * HDIM, nullptr);
    // s_v_hat = tanh( [H_v | gH_c] @ [Ws2;Ws3] )
    wmma_gemm_bf16<2, 0, 1, false><<<gShat, blk, 0, stream>>>(
        Hv, HDIM, HDIM, Hc, HDIM, HDIM, BtWs23, nullptr, shat, HDIM, gvec);
    update_v_kernel<<<gUpd, blk, 0, stream>>>(gates, shat, gvec, svs, Hv);

    // H_v_p = predict_v(H_v) * H_c_p  -> pred_v[t]
    predict(Hv, BtPv0, bpv0, BtPv1, bpv1, Wfcv, bfcv,
            gvec, nullptr, pred_v + (long)t * BATCH);
  }
}
